// GraphEditModel_64415919505849
// MI455X (gfx1250) — matrix operand order repaired and checked by reading, not compiled, not run on hardware
//
#include <hip/hip_runtime.h>
#include <hip/hip_bf16.h>
#include <stdint.h>

#define N_NODES    262144
#define NODE_DIM   128
#define EDGE_DIM   128
#define EDGE_CLASS 5
#define N_EDGES    1048576

#define LDK     136      // padded row stride in halves (272B -> 4-bank skew/row, conflict-free b128)
#define TILE_M  128      // edges per block
#define THREADS 256      // 8 wave32 waves

typedef __attribute__((ext_vector_type(16))) _Float16     v16h;
typedef __attribute__((ext_vector_type(8)))  float        v8f;
typedef __attribute__((ext_vector_type(4)))  unsigned int v4u;
typedef __attribute__((ext_vector_type(2)))  unsigned int v2u;

union Fragu { v16h h; v4u u[2]; };
union Pack4 { _Float16 h[4]; v2u u; };

__device__ __forceinline__ v8f splat8(float x) {
  v8f v = {x, x, x, x, x, x, x, x};
  return v;
}

// A fragment (16x32 f16, row-major buf, stride LDK):
//   lane holds row (lane&15); halves 0..7 -> k = kt*32 + g*8 + j; halves 8..15 -> +16
__device__ __forceinline__ v16h load_frag_a(const _Float16* __restrict__ buf,
                                            int row, int g, int kt) {
  const _Float16* p = buf + row * LDK + kt * 32 + g * 8;
  Fragu f;
  f.u[0] = *(const v4u*)p;
  f.u[1] = *(const v4u*)(p + 16);
  return f.h;
}

// B fragment (32x16 f16) from transposed weights WT[n][k] (stride LDK):
//   lane holds col n = (lane&15); halves 0..7 -> k = kt*32 + g*8 + j; halves 8..15 -> +16
__device__ __forceinline__ v16h load_frag_b(const _Float16* __restrict__ buf,
                                            int col, int g, int kt) {
  const _Float16* p = buf + col * LDK + kt * 32 + g * 8;
  Fragu f;
  f.u[0] = *(const v4u*)p;
  f.u[1] = *(const v4u*)(p + 16);
  return f.h;
}

// ---------------------------------------------------------------------------
// Prep: fold Wsum = Wa + Wb, transpose weights to f16 WT[n][k] (padded stride),
// pad W2 to N=16, pack biases (2*b_agg, b1, b2 padded). 64 blocks x 256 thr.
// ---------------------------------------------------------------------------
__global__ __launch_bounds__(256)
void prep_weights(const float* __restrict__ W_agg, const float* __restrict__ b_agg,
                  const float* __restrict__ W1,    const float* __restrict__ b1,
                  const float* __restrict__ W2,    const float* __restrict__ b2,
                  _Float16* __restrict__ wpack,    float* __restrict__ bpack) {
  const int idx = blockIdx.x * 256 + threadIdx.x;   // 0..16383
  const int n = idx & 127;
  const int k = idx >> 7;
  _Float16* WsumT = wpack;                 // 128 x LDK
  _Float16* W1T   = wpack + 128 * LDK;     // 128 x LDK
  _Float16* W2T   = W1T   + 128 * LDK;     // 16  x LDK

  WsumT[n * LDK + k] = (_Float16)(W_agg[k * 128 + n] + W_agg[(k + 128) * 128 + n]);
  W1T[n * LDK + k]   = (_Float16)W1[k * 128 + n];
  if (n < 16)
    W2T[n * LDK + k] = (_Float16)((n < EDGE_CLASS) ? W2[k * EDGE_CLASS + n] : 0.0f);
  if (idx < 128) {
    bpack[idx]       = 2.0f * b_agg[idx];
    bpack[128 + idx] = b1[idx];
  }
  if (idx < 16)
    bpack[256 + idx] = (idx < EDGE_CLASS) ? b2[idx] : 0.0f;
}

// ---------------------------------------------------------------------------
// Fused edge MLP: gather+sum -> GEMM(128x128x128) -> relu GEMM -> GEMM(Npad16)
// 128 edges per block; each wave owns a 16-row strip end-to-end.
// ---------------------------------------------------------------------------
__global__ __launch_bounds__(THREADS)
void edge_mlp_fused(const float* __restrict__ node_feat,
                    const long long* __restrict__ edge_index,
                    const _Float16* __restrict__ wpack,
                    const float* __restrict__ bpack,
                    float* __restrict__ out) {
  extern __shared__ char smem[];
  _Float16* sX    = (_Float16*)smem;            // TILE_M x LDK (X -> E -> H in place)
  _Float16* sWsum = sX    + TILE_M * LDK;       // 128 x LDK
  _Float16* sW1   = sWsum + 128 * LDK;          // 128 x LDK
  _Float16* sW2   = sW1   + 128 * LDK;          // 16  x LDK
  float*    sB    = (float*)(sW2 + 16 * LDK);   // [0,128): 2*b_agg  [128,256): b1  [256,272): b2

  const int tid  = threadIdx.x;
  const int lane = tid & 31;
  const int g    = lane >> 4;
  const int ln16 = lane & 15;
  const int wv   = tid >> 5;

  // ---- stage precomputed f16 weights + biases into LDS ----
  {
    const v4u* src = (const v4u*)wpack;
    v4u* dst = (v4u*)sWsum;
    const int nvec = (2 * 128 * LDK + 16 * LDK) * 2 / 16;   // 4624 x 16B
    for (int i = tid; i < nvec; i += THREADS) dst[i] = src[i];
    for (int i = tid; i < 272; i += THREADS) sB[i] = bpack[i];
  }

  // ---- gather: x = node[src] + node[dst] (f32), convert to f16 into sX ----
  {
    const int e_local = tid >> 1;              // 2 threads per edge
    const int half    = tid & 1;               // 64-feature half
    const long long e  = (long long)blockIdx.x * TILE_M + e_local;
    const long long si = edge_index[e];
    const long long di = edge_index[(long long)N_EDGES + e];
    const float4* sp = (const float4*)(node_feat + si * NODE_DIM) + half * 16;
    const float4* dp = (const float4*)(node_feat + di * NODE_DIM) + half * 16;
    v2u* xp = (v2u*)(sX + e_local * LDK + half * 64);
#pragma unroll
    for (int j = 0; j < 16; ++j) {
      float4 a = sp[j];
      float4 b = dp[j];
      Pack4 pk;
      pk.h[0] = (_Float16)(a.x + b.x);
      pk.h[1] = (_Float16)(a.y + b.y);
      pk.h[2] = (_Float16)(a.z + b.z);
      pk.h[3] = (_Float16)(a.w + b.w);
      xp[j] = pk.u;
    }
  }
  __syncthreads();   // weights staged cooperatively; after this everything is wave-local

  const int rbase = wv * 16;          // this wave's 16 rows in the 128-edge tile
  const int row   = rbase + ln16;

  v8f acc[8];

  // ---------- stage 1: E = X @ Wsum + 2*b_agg ----------
#pragma unroll
  for (int n = 0; n < 8; ++n) acc[n] = splat8(sB[n * 16 + ln16]);
#pragma unroll
  for (int kt = 0; kt < 4; ++kt) {
    v16h a = load_frag_a(sX, row, g, kt);
#pragma unroll
    for (int n = 0; n < 8; ++n) {
      v16h b = load_frag_b(sWsum, n * 16 + ln16, g, kt);
      acc[n] = __builtin_amdgcn_wmma_f32_16x16x32_f16(false, a, false, b,
                                                      (short)0, acc[n], false, false);
    }
  }
  // write E (f16) back over this wave's rows of sX (all reads of X already done)
#pragma unroll
  for (int n = 0; n < 8; ++n)
#pragma unroll
    for (int v = 0; v < 8; ++v)
      sX[(rbase + v + 8 * g) * LDK + n * 16 + ln16] = (_Float16)acc[n][v];

  // ---------- stage 2: H = relu(E @ W1 + b1) ----------
#pragma unroll
  for (int n = 0; n < 8; ++n) acc[n] = splat8(sB[128 + n * 16 + ln16]);
#pragma unroll
  for (int kt = 0; kt < 4; ++kt) {
    v16h a = load_frag_a(sX, row, g, kt);
#pragma unroll
    for (int n = 0; n < 8; ++n) {
      v16h b = load_frag_b(sW1, n * 16 + ln16, g, kt);
      acc[n] = __builtin_amdgcn_wmma_f32_16x16x32_f16(false, a, false, b,
                                                      (short)0, acc[n], false, false);
    }
  }
#pragma unroll
  for (int n = 0; n < 8; ++n)
#pragma unroll
    for (int v = 0; v < 8; ++v)
      sX[(rbase + v + 8 * g) * LDK + n * 16 + ln16] =
          (_Float16)fmaxf(acc[n][v], 0.0f);

  // ---------- stage 3: OUT = H @ W2 + b2 (N padded to 16, keep 5) ----------
  v8f c3 = splat8(sB[256 + ln16]);
#pragma unroll
  for (int kt = 0; kt < 4; ++kt) {
    v16h a = load_frag_a(sX, row, g, kt);
    v16h b = load_frag_b(sW2, ln16, g, kt);
    c3 = __builtin_amdgcn_wmma_f32_16x16x32_f16(false, a, false, b,
                                                (short)0, c3, false, false);
  }
  if (ln16 < EDGE_CLASS) {
    const long long ebase = (long long)blockIdx.x * TILE_M + rbase;
#pragma unroll
    for (int v = 0; v < 8; ++v)
      out[(ebase + v + 8 * g) * EDGE_CLASS + ln16] = c3[v];
  }
}

extern "C" void kernel_launch(void* const* d_in, const int* in_sizes, int n_in,
                              void* d_out, int out_size, void* d_ws, size_t ws_size,
                              hipStream_t stream) {
  (void)in_sizes; (void)n_in; (void)out_size; (void)ws_size;
  const float*     node_feat  = (const float*)d_in[0];
  const long long* edge_index = (const long long*)d_in[1];
  const float*     W_agg      = (const float*)d_in[2];
  const float*     b_agg      = (const float*)d_in[3];
  const float*     W1         = (const float*)d_in[4];
  const float*     b1         = (const float*)d_in[5];
  const float*     W2         = (const float*)d_in[6];
  const float*     b2         = (const float*)d_in[7];

  // workspace layout: [WsumT | W1T | W2T] f16 (padded stride), then biases f32
  const size_t wpack_bytes = (size_t)(2 * 128 * LDK + 16 * LDK) * sizeof(_Float16); // 73984
  _Float16* wpack = (_Float16*)d_ws;
  float*    bpack = (float*)((char*)d_ws + wpack_bytes);

  prep_weights<<<64, 256, 0, stream>>>(W_agg, b_agg, W1, b1, W2, b2, wpack, bpack);

  const size_t smem_bytes =
      (size_t)(TILE_M * LDK + 2 * 128 * LDK + 16 * LDK) * sizeof(_Float16) + 272 * sizeof(float);
  // 109,888 B per block (<= 320 KB/WGP, 2 blocks/WGP) — dynamic LDS, legal on CDNA5
  edge_mlp_fused<<<N_EDGES / TILE_M, THREADS, smem_bytes, stream>>>(
      node_feat, edge_index, wpack, bpack, (float*)d_out);
}
